// GCN_44702019616959
// MI455X (gfx1250) — matrix operand order repaired and checked by reading, not compiled, not run on hardware
//
#include <hip/hip_runtime.h>
#include <hip/hip_bf16.h>

typedef _Float16 v16h __attribute__((ext_vector_type(16)));
typedef float    v8f  __attribute__((ext_vector_type(8)));

#define IN_DIM   128
#define OUT_DIM  32
#define N_HEADS  4
#define OC       128   // N_HEADS * OUT_DIM
#define NEG_SLOPE 0.01f

// ---------------------------------------------------------------------------
// K0: repack W [H,O,I] f32 -> f16 B-fragments laid out exactly per-lane:
// frag index (kt, nt), lane, 16 halves. For 16-bit B (32x16, wave32):
//   lanes 0-15 hold K=0..7,16..23 (halves j=0..15), lanes 16-31 hold K=8..15,24..31.
//   N = nt*16 + (lane&15), Kglobal = kt*32 + (lane>>4)*8 + (j&7) + (j>>3)*16.
// Wt[K][N] = W[(N)*128 + K]  (since col N = h*32+o and W flat is [(h*32+o)*128+i])
// ---------------------------------------------------------------------------
__global__ void gat_repack_w(const float* __restrict__ W, _Float16* __restrict__ wp) {
    int t = blockIdx.x * blockDim.x + threadIdx.x;      // 16384 threads
    if (t >= (OC / 16) * (IN_DIM / 32) * 32 * 16) return;
    int j    = t & 15;
    int lane = (t >> 4) & 31;
    int nt   = (t >> 9) & 7;
    int kt   = t >> 12;
    int Ng   = nt * 16 + (lane & 15);
    int Kg   = kt * 32 + ((lane >> 4) << 3) + (j & 7) + ((j >> 3) << 4);
    wp[t] = (_Float16)W[(size_t)Ng * IN_DIM + Kg];
}

// ---------------------------------------------------------------------------
// K1: z[N,128] = h[N,128] @ Wt[128,128] via v_wmma_f32_16x16x32_f16.
// One wave -> 16 rows x 128 cols (8 n-tiles, 4 k-steps => 32 WMMA).
// ---------------------------------------------------------------------------
#define WAVES_PER_BLOCK 4
__global__ void gat_gemm(const float* __restrict__ h,
                         const _Float16* __restrict__ wp,
                         float* __restrict__ z, int Nn) {
    int wave = threadIdx.x >> 5;
    int lane = threadIdx.x & 31;
    int half = lane >> 4;                 // 0 or 1
    int m0 = (blockIdx.x * WAVES_PER_BLOCK + wave) * 16;
    if (m0 >= Nn) return;                 // wave-uniform exit; EXEC all-ones below

    int r = m0 + (lane & 15);
    if (r >= Nn) r = Nn - 1;              // clamp loads; stores guarded later
    const float* hrow = h + (size_t)r * IN_DIM;
    const v16h* bfrag = (const v16h*)wp;

    v8f acc[8];
#pragma unroll
    for (int nt = 0; nt < 8; ++nt) acc[nt] = (v8f)0.0f;

#pragma unroll
    for (int kt = 0; kt < 4; ++kt) {
        int base = kt * 32 + half * 8;
        float4 fa = *(const float4*)(hrow + base);
        float4 fb = *(const float4*)(hrow + base + 4);
        float4 fc = *(const float4*)(hrow + base + 16);
        float4 fd = *(const float4*)(hrow + base + 20);
        v16h A;
        A[0]=(_Float16)fa.x; A[1]=(_Float16)fa.y; A[2]=(_Float16)fa.z; A[3]=(_Float16)fa.w;
        A[4]=(_Float16)fb.x; A[5]=(_Float16)fb.y; A[6]=(_Float16)fb.z; A[7]=(_Float16)fb.w;
        A[8]=(_Float16)fc.x; A[9]=(_Float16)fc.y; A[10]=(_Float16)fc.z; A[11]=(_Float16)fc.w;
        A[12]=(_Float16)fd.x; A[13]=(_Float16)fd.y; A[14]=(_Float16)fd.z; A[15]=(_Float16)fd.w;
#pragma unroll
        for (int nt = 0; nt < 8; ++nt) {
            v16h B = bfrag[(kt * 8 + nt) * 32 + lane];
            acc[nt] = __builtin_amdgcn_wmma_f32_16x16x32_f16(
                false, A, false, B, (short)0, acc[nt], false, false);
        }
    }

    // C/D layout: VGPR r -> M = m0 + r + half*8 ; N = nt*16 + (lane&15)
#pragma unroll
    for (int nt = 0; nt < 8; ++nt) {
#pragma unroll
        for (int rr = 0; rr < 8; ++rr) {
            int row = m0 + rr + half * 8;
            if (row < Nn) z[(size_t)row * OC + nt * 16 + (lane & 15)] = acc[nt][rr];
        }
    }
}

// ---------------------------------------------------------------------------
// K2: per-node attention partial dots  s_src[n,h], s_dst[n,h]
// ---------------------------------------------------------------------------
__global__ void gat_attn_dots(const float* __restrict__ z, const float* __restrict__ a,
                              float* __restrict__ s_src, float* __restrict__ s_dst, int Nn) {
    int t = blockIdx.x * blockDim.x + threadIdx.x;
    if (t >= Nn * N_HEADS) return;
    int n = t >> 2, hd = t & 3;
    const float* zr = z + (size_t)n * OC + hd * OUT_DIM;
    const float* ar = a + hd * (2 * OUT_DIM);
    float s1 = 0.f, s2 = 0.f;
#pragma unroll
    for (int j = 0; j < OUT_DIM; ++j) {
        float zv = zr[j];
        s1 += zv * ar[j];
        s2 += zv * ar[OUT_DIM + j];
    }
    s_src[t] = s1;
    s_dst[t] = s2;
}

// ---------------------------------------------------------------------------
// K3: init out=0, nodemax=-inf (bits), denom=0
// ---------------------------------------------------------------------------
__global__ void gat_init(float* __restrict__ out, unsigned* __restrict__ nmax_u,
                         float* __restrict__ den, int Nn) {
    int i = blockIdx.x * blockDim.x + threadIdx.x;
    if (i < Nn * OC) out[i] = 0.f;
    if (i < Nn * N_HEADS) { nmax_u[i] = 0xFF800000u; den[i] = 0.f; }
}

__device__ inline void atomicMaxF32(float* addr, float v) {
    if (v >= 0.f) atomicMax((int*)addr, __float_as_int(v));
    else          atomicMin((unsigned int*)addr, (unsigned int)__float_as_int(v));
}

// ---------------------------------------------------------------------------
// K4: e = leaky(s_src[src]+s_dst[dst]); atomic segment-max over dst
// ---------------------------------------------------------------------------
__global__ void gat_edge_logits(const int* __restrict__ src, const int* __restrict__ dst,
                                const float* __restrict__ s_src, const float* __restrict__ s_dst,
                                float* __restrict__ ebuf, float* __restrict__ nmax, int E) {
    int e = blockIdx.x * blockDim.x + threadIdx.x;
    if (e >= E) return;
    int s = src[e], d = dst[e];
    float4 ss = *(const float4*)(s_src + (size_t)s * 4);
    float4 sd = *(const float4*)(s_dst + (size_t)d * 4);
    float ev[4] = { ss.x + sd.x, ss.y + sd.y, ss.z + sd.z, ss.w + sd.w };
#pragma unroll
    for (int hh = 0; hh < 4; ++hh) {
        ev[hh] = ev[hh] >= 0.f ? ev[hh] : NEG_SLOPE * ev[hh];
        atomicMaxF32(&nmax[(size_t)d * 4 + hh], ev[hh]);
    }
    *(float4*)(ebuf + (size_t)e * 4) = make_float4(ev[0], ev[1], ev[2], ev[3]);
}

// ---------------------------------------------------------------------------
// K5: ex = exp(e - max[dst]); denom[dst] += ex
// ---------------------------------------------------------------------------
__global__ void gat_edge_exp(const int* __restrict__ dst, float* __restrict__ ebuf,
                             const float* __restrict__ nmax, float* __restrict__ den, int E) {
    int e = blockIdx.x * blockDim.x + threadIdx.x;
    if (e >= E) return;
    int d = dst[e];
    float4 ev = *(const float4*)(ebuf + (size_t)e * 4);
    float4 mx = *(const float4*)(nmax + (size_t)d * 4);
    float ex[4] = { __expf(ev.x - mx.x), __expf(ev.y - mx.y),
                    __expf(ev.z - mx.z), __expf(ev.w - mx.w) };
#pragma unroll
    for (int hh = 0; hh < 4; ++hh) atomicAdd(&den[(size_t)d * 4 + hh], ex[hh]);
    *(float4*)(ebuf + (size_t)e * 4) = make_float4(ex[0], ex[1], ex[2], ex[3]);
}

// ---------------------------------------------------------------------------
// K6: one wave per edge: out[dst,:] += alpha * z[src,:]
// lane covers 4 consecutive columns; head = lane>>3. z & out are L2-resident.
// ---------------------------------------------------------------------------
__global__ void gat_scatter(const int* __restrict__ src, const int* __restrict__ dst,
                            const float* __restrict__ z, const float* __restrict__ ebuf,
                            const float* __restrict__ den, float* __restrict__ out, int E) {
    int t = blockIdx.x * blockDim.x + threadIdx.x;
    int e = t >> 5;
    if (e >= E) return;
    int lane = t & 31;
    int s = src[e], d = dst[e];
    int c0 = lane * 4;
    int hd = lane >> 3;
    float dn = den[(size_t)d * 4 + hd];
    float alpha = ebuf[(size_t)e * 4 + hd] / (dn > 0.f ? dn : 1.f);
    float4 zz = *(const float4*)(z + (size_t)s * OC + c0);
    float* o = out + (size_t)d * OC + c0;
    atomicAdd(o + 0, alpha * zz.x);
    atomicAdd(o + 1, alpha * zz.y);
    atomicAdd(o + 2, alpha * zz.z);
    atomicAdd(o + 3, alpha * zz.w);
}

// ---------------------------------------------------------------------------
extern "C" void kernel_launch(void* const* d_in, const int* in_sizes, int n_in,
                              void* d_out, int out_size, void* d_ws, size_t ws_size,
                              hipStream_t stream) {
    const float* h   = (const float*)d_in[0];
    const float* W   = (const float*)d_in[1];
    const float* a   = (const float*)d_in[2];
    const int*   src = (const int*)d_in[3];
    const int*   dst = (const int*)d_in[4];
    float* out = (float*)d_out;

    int Nn = in_sizes[0] / IN_DIM;
    int E  = in_sizes[3];

    // workspace carve-up (z | s_src | s_dst | nmax | den | ebuf | wpack)
    float* z     = (float*)d_ws;
    float* s_src = z + (size_t)Nn * OC;
    float* s_dst = s_src + (size_t)Nn * N_HEADS;
    float* nmax  = s_dst + (size_t)Nn * N_HEADS;
    float* den   = nmax + (size_t)Nn * N_HEADS;
    float* ebuf  = den + (size_t)Nn * N_HEADS;
    uintptr_t wp_addr = (uintptr_t)(ebuf + (size_t)E * N_HEADS);
    wp_addr = (wp_addr + 31) & ~(uintptr_t)31;          // 32B align for v16h loads
    _Float16* wpack = (_Float16*)wp_addr;

    // K0: repack W into per-lane B fragments (16384 halves)
    gat_repack_w<<<64, 256, 0, stream>>>(W, wpack);

    // K1: projection GEMM via WMMA (wave = 16 rows x 128 cols)
    int rows_per_block = WAVES_PER_BLOCK * 16;
    int gemm_blocks = (Nn + rows_per_block - 1) / rows_per_block;
    gat_gemm<<<gemm_blocks, WAVES_PER_BLOCK * 32, 0, stream>>>(h, wpack, z, Nn);

    // K2: per-node attention dots
    int nt = Nn * N_HEADS;
    gat_attn_dots<<<(nt + 255) / 256, 256, 0, stream>>>(z, a, s_src, s_dst, Nn);

    // K3: init out / nodemax / denom
    int ni = Nn * OC;
    gat_init<<<(ni + 255) / 256, 256, 0, stream>>>(out, (unsigned*)nmax, den, Nn);

    // K4-K6: edge softmax + weighted scatter (L2-resident gather/scatter)
    gat_edge_logits<<<(E + 255) / 256, 256, 0, stream>>>(src, dst, s_src, s_dst, ebuf, nmax, E);
    gat_edge_exp<<<(E + 255) / 256, 256, 0, stream>>>(dst, ebuf, nmax, den, E);
    long long st = (long long)E * 32;
    gat_scatter<<<(int)((st + 255) / 256), 256, 0, stream>>>(src, dst, z, ebuf, den, out, E);
}
